// TransformerEncoderReadout_mol_10222022164948
// MI455X (gfx1250) — compile-verified, matrix operand order checked by reading, not run on hardware
//
#include <hip/hip_runtime.h>
#include <hip/hip_bf16.h>

typedef __attribute__((ext_vector_type(16))) _Float16 v16h;
typedef __attribute__((ext_vector_type(8)))  float    v8f;
typedef __attribute__((ext_vector_type(8)))  _Float16 h8;
typedef __attribute__((ext_vector_type(4)))  _Float16 h4;

#define D_EMB   128
#define HKD     1024     // H * KD
#define KV_W    2048     // [K | V] packed per atom
#define LN_EPS  1e-3f
#define QSCALE  0.08838834764831845f   // 1/sqrt(128)

#define KB   64          // K-depth per LDS stage
#define SAH  72          // halves per LDS row (64 + 8 pad)
#define SAW  36          // dwords per LDS row
#define SOH  136         // halves per LDS row for the D-staging buffer

union Frag { unsigned int u[8]; v16h h; };

// ---------------------------------------------------------------------------
// Kernel 0: segment offsets from sorted molecule ids (no atomics)
// ---------------------------------------------------------------------------
__global__ void build_offsets(const int* __restrict__ mol,
                              int* __restrict__ offsets, int N, int B) {
    int i = blockIdx.x * blockDim.x + threadIdx.x;
    if (i == 0) { offsets[mol[0]] = 0; offsets[B] = N; }
    if (i > 0 && i < N) {
        int m = mol[i];
        if (m != mol[i - 1]) offsets[m] = i;
    }
}

// ---------------------------------------------------------------------------
// Kernel 1: KV = atom_features @ [Wk | Wv] + [bk | bv], stored f16 [Npad,2048]
// WMMA f32_16x16x32_f16; block tile 128(M) x 128(N), K=128 in two 64-deep
// LDS stages. 8 waves: wave w owns rows [16w,16w+16), all 128 cols.
// ---------------------------------------------------------------------------
__global__ __launch_bounds__(256) void kv_gemm(
    const float* __restrict__ feat, const float* __restrict__ Wk,
    const float* __restrict__ bk,   const float* __restrict__ Wv,
    const float* __restrict__ bv,   _Float16* __restrict__ kv, int N)
{
    __shared__ _Float16 smem[2 * 128 * SAH];     // 36 KB
    _Float16* sA  = smem;                        // A tile: rows x k
    _Float16* sBT = smem + 128 * SAH;            // B tile transposed: col x k

    const int tid  = threadIdx.x;
    const long long i0 = (long long)blockIdx.x * 128;
    const int jblk = blockIdx.y;                       // 0..15 (8 K-cols, 8 V-cols)
    const float* __restrict__ srcW = (jblk < 8) ? Wk : Wv;
    const float* __restrict__ srcB = (jblk < 8) ? bk : bv;
    const int js0 = (jblk & 7) * 128;                  // col offset within [0,1024)
    const bool fullM = (i0 + 128) <= (long long)N;     // uniform per block

    const int wave = tid >> 5, lane = tid & 31;
    const int lm = lane & 15, kg = lane >> 4;
    const int m0 = wave * 16;

    v8f acc[8];
#pragma unroll
    for (int nt = 0; nt < 8; ++nt) {
        v8f z;
#pragma unroll
        for (int e = 0; e < 8; ++e) z[e] = 0.f;
        acc[nt] = z;
    }

    const unsigned int* A32 = (const unsigned int*)sA;
    const unsigned int* B32 = (const unsigned int*)sBT;

    for (int kb = 0; kb < 2; ++kb) {
        // ---- A tile: 128 rows x 64 k ; float4 loads -> h4 stores ----
        if (fullM) {
            float4 av[8];
#pragma unroll
            for (int e = 0; e < 8; ++e) {
                int qi = e * 256 + tid;                 // 0..2047 quads
                int r = qi >> 4, c4 = (qi & 15) << 2;
                av[e] = *(const float4*)(feat + (i0 + r) * D_EMB + kb * KB + c4);
            }
#pragma unroll
            for (int e = 0; e < 8; ++e) {
                int qi = e * 256 + tid;
                int r = qi >> 4, c4 = (qi & 15) << 2;
                h4 hv;
                hv[0] = (_Float16)av[e].x; hv[1] = (_Float16)av[e].y;
                hv[2] = (_Float16)av[e].z; hv[3] = (_Float16)av[e].w;
                *(h4*)(sA + r * SAH + c4) = hv;
            }
        } else {
#pragma unroll
            for (int e = 0; e < 8; ++e) {
                int qi = e * 256 + tid;
                int r = qi >> 4, c4 = (qi & 15) << 2;
                float4 v = make_float4(0.f, 0.f, 0.f, 0.f);
                if ((i0 + r) < N)
                    v = *(const float4*)(feat + (i0 + r) * D_EMB + kb * KB + c4);
                h4 hv;
                hv[0] = (_Float16)v.x; hv[1] = (_Float16)v.y;
                hv[2] = (_Float16)v.z; hv[3] = (_Float16)v.w;
                *(h4*)(sA + r * SAH + c4) = hv;
            }
        }

        // ---- B tile: 64 k x 128 cols, transposed store; 4 coalesced row
        //      loads -> one h4 store along k ----
        {
            float bw[8][4];
#pragma unroll
            for (int e = 0; e < 8; ++e) {
                int qi = e * 256 + tid;                 // 0..2047
                int jj = qi & 127;
                int dd0 = (qi >> 7) << 2;               // 0,4,...,60
                const float* src = srcW + (long long)(kb * KB + dd0) * HKD + js0 + jj;
#pragma unroll
                for (int i = 0; i < 4; ++i) bw[e][i] = src[(long long)i * HKD];
            }
#pragma unroll
            for (int e = 0; e < 8; ++e) {
                int qi = e * 256 + tid;
                int jj = qi & 127;
                int dd0 = (qi >> 7) << 2;
                h4 hv;
#pragma unroll
                for (int i = 0; i < 4; ++i) hv[i] = (_Float16)bw[e][i];
                *(h4*)(sBT + jj * SAH + dd0) = hv;
            }
        }
        __syncthreads();

#pragma unroll
        for (int ks = 0; ks < 2; ++ks) {
            // A fragment (16x32 f16): lane lm = row, kg selects K half-group
            Frag a;
            const int abase = (m0 + lm) * SAW + ks * 16 + kg * 4;
#pragma unroll
            for (int v = 0; v < 8; ++v)
                a.u[v] = A32[abase + (v & 3) + ((v >> 2) << 3)];
#pragma unroll
            for (int nt = 0; nt < 8; ++nt) {
                Frag b;   // B fragment: same layout against transposed tile
                const int bbase = (nt * 16 + lm) * SAW + ks * 16 + kg * 4;
#pragma unroll
                for (int v = 0; v < 8; ++v)
                    b.u[v] = B32[bbase + (v & 3) + ((v >> 2) << 3)];
                acc[nt] = __builtin_amdgcn_wmma_f32_16x16x32_f16(
                    false, a.h, false, b.h, (short)0, acc[nt], false, false);
            }
        }
        __syncthreads();
    }

    // ---- epilogue: +bias, stage D in LDS, then coalesced 16B stores ----
    _Float16* sOut = smem;                       // 128 x SOH halves (fits)
#pragma unroll
    for (int nt = 0; nt < 8; ++nt) {
        const float bias = srcB[js0 + nt * 16 + lm];
#pragma unroll
        for (int v = 0; v < 8; ++v) {
            int row = m0 + (kg ? v + 8 : v);
            sOut[row * SOH + nt * 16 + lm] = (_Float16)(acc[nt][v] + bias);
        }
    }
    __syncthreads();
#pragma unroll
    for (int e = 0; e < 8; ++e) {
        int qi = e * 256 + tid;                  // 0..2047 octets
        int row = qi >> 4, c8 = (qi & 15) << 3;
        h8 val = *(const h8*)(sOut + row * SOH + c8);
        *(h8*)(kv + (i0 + row) * KV_W + jblk * 128 + c8) = val;
    }
}

// ---------------------------------------------------------------------------
// block-wide sum over provided values (all 256 threads participate)
// ---------------------------------------------------------------------------
__device__ __forceinline__ float block_sum(float v, float* red, int t) {
    red[t] = v;
    __syncthreads();
#pragma unroll
    for (int s = 128; s > 0; s >>= 1) {
        if (t < s) red[t] = red[t] + red[t + s];
        __syncthreads();
    }
    float r = red[0];
    __syncthreads();
    return r;
}

// ---------------------------------------------------------------------------
// Kernel 2: per-molecule fused tail. Only the picked row of everything after
// attention is needed; only the picked query row of attention is needed.
// ---------------------------------------------------------------------------
__global__ __launch_bounds__(256) void attn_tail(
    const float* __restrict__ feat, const float* __restrict__ Wq,
    const float* __restrict__ bq,   const float* __restrict__ Wo,
    const float* __restrict__ bo,   const float* __restrict__ W1,
    const float* __restrict__ b1,   const float* __restrict__ W2,
    const float* __restrict__ b2,   const float* __restrict__ g1,
    const float* __restrict__ be1,  const float* __restrict__ g2,
    const float* __restrict__ be2,  const int* __restrict__ offsets,
    const int* __restrict__ picked, const _Float16* __restrict__ kv,
    float* __restrict__ out, int N)
{
    __shared__ float xp[128];
    __shared__ float q[1024];
    __shared__ float sc[1024];   // [h][s] scores, then attn probs in place
    __shared__ float ctx[1024];
    __shared__ float h1[128];
    __shared__ float Fb[512];
    __shared__ float red[256];

    const int b = blockIdx.x, t = threadIdx.x;
    const int o0 = offsets[b];
    const int count = offsets[b + 1] - o0;
    const long long g = (long long)o0 + picked[b];

    if (t < 128) xp[t] = feat[g * D_EMB + t];
    __syncthreads();

    // q = (x_p @ Wq + bq) / sqrt(KD)   — 4 outputs per thread
    {
        float a0 = 0.f, a1 = 0.f, a2 = 0.f, a3 = 0.f;
        for (int d = 0; d < 128; ++d) {
            float xv = xp[d];
            const float* wr = Wq + (long long)d * HKD;
            a0 += xv * wr[t];       a1 += xv * wr[t + 256];
            a2 += xv * wr[t + 512]; a3 += xv * wr[t + 768];
        }
        q[t]       = (a0 + bq[t])       * QSCALE;
        q[t + 256] = (a1 + bq[t + 256]) * QSCALE;
        q[t + 512] = (a2 + bq[t + 512]) * QSCALE;
        q[t + 768] = (a3 + bq[t + 768]) * QSCALE;
    }
    __syncthreads();

    // scores[h][s] = q_h . K[o0+s]_h   (vectorized 16B f16 loads)
#pragma unroll
    for (int c = 0; c < 4; ++c) {
        int p = t + 256 * c;
        int h = p >> 7, s = p & 127;
        float a = 0.f;
        if (s < count) {
            const h8* kp = (const h8*)(kv + (long long)(o0 + s) * KV_W + h * 128);
            const float* qh = q + h * 128;
#pragma unroll
            for (int u = 0; u < 16; ++u) {
                h8 kk = kp[u];
#pragma unroll
                for (int e = 0; e < 8; ++e) a += qh[u * 8 + e] * (float)kk[e];
            }
        }
        sc[p] = a;
    }
    __syncthreads();

    // softmax over valid keys: one wave (32 lanes) per head, shfl reductions
    {
        int w = t >> 5, lane = t & 31;
        float* row = sc + w * 128;
        float m = -3.0e38f;
        for (int s = lane; s < count; s += 32) m = fmaxf(m, row[s]);
#pragma unroll
        for (int off = 16; off > 0; off >>= 1) m = fmaxf(m, __shfl_xor(m, off));
        float sum = 0.f;
        for (int s = lane; s < count; s += 32) {
            float e = __expf(row[s] - m);
            row[s] = e;
            sum += e;
        }
#pragma unroll
        for (int off = 16; off > 0; off >>= 1) sum += __shfl_xor(sum, off);
        float inv = 1.f / sum;
        for (int s = lane; s < count; s += 32) row[s] *= inv;
    }
    __syncthreads();

    // ctx[h][k] = sum_s attn[h][s] * V[o0+s][h][k]
#pragma unroll
    for (int c = 0; c < 4; ++c) {
        int j = t + 256 * c;
        int h = j >> 7;
        float a = 0.f;
        const float* arow = sc + h * 128;
        for (int s = 0; s < count; ++s)
            a += arow[s] * (float)kv[(long long)(o0 + s) * KV_W + HKD + j];
        ctx[j] = a;
    }
    __syncthreads();

    // attn_out row: split the 1024-deep GEMV across both thread halves
    {
        int d = t & 127, half = t >> 7;
        float part = 0.f;
        const float* cp = ctx + half * 512;
        const float* wp = Wo + (long long)(half * 512) * 128 + d;
        for (int j = 0; j < 512; ++j) part += cp[j] * wp[(long long)j * 128];
        red[t] = part;
    }
    __syncthreads();
    float r1 = 0.f;
    if (t < 128) r1 = xp[t] + bo[t] + red[t] + red[t + 128];
    __syncthreads();

    // LayerNorm 1
    float sum  = block_sum(t < 128 ? r1      : 0.f, red, t);
    float sq   = block_sum(t < 128 ? r1 * r1 : 0.f, red, t);
    float mean = sum * (1.f / 128.f);
    float var  = sq  * (1.f / 128.f) - mean * mean;
    float rstd = rsqrtf(var + LN_EPS);
    if (t < 128) h1[t] = (r1 - mean) * rstd * g1[t] + be1[t];
    __syncthreads();

    // FFN layer 1 + ReLU (2 outputs / thread)
#pragma unroll
    for (int c = 0; c < 2; ++c) {
        int f = t + 256 * c;
        float a = b1[f];
        for (int d = 0; d < 128; ++d) a += h1[d] * W1[(long long)d * 512 + f];
        Fb[f] = fmaxf(a, 0.f);
    }
    __syncthreads();

    // FFN layer 2: split the 512-deep GEMV across both thread halves
    {
        int d = t & 127, half = t >> 7;
        float part = 0.f;
        const float* fp = Fb + half * 256;
        const float* wp = W2 + (long long)(half * 256) * 128 + d;
        for (int f = 0; f < 256; ++f) part += fp[f] * wp[(long long)f * 128];
        red[t] = part;
    }
    __syncthreads();
    float r2 = 0.f;
    if (t < 128) r2 = h1[t] + b2[t] + red[t] + red[t + 128];
    __syncthreads();

    // LayerNorm 2 -> picked-row output
    float sum2  = block_sum(t < 128 ? r2      : 0.f, red, t);
    float sq2   = block_sum(t < 128 ? r2 * r2 : 0.f, red, t);
    float mean2 = sum2 * (1.f / 128.f);
    float var2  = sq2  * (1.f / 128.f) - mean2 * mean2;
    float rstd2 = rsqrtf(var2 + LN_EPS);
    if (t < 128) out[(long long)b * 128 + t] = (r2 - mean2) * rstd2 * g2[t] + be2[t];
}

// ---------------------------------------------------------------------------
extern "C" void kernel_launch(void* const* d_in, const int* in_sizes, int n_in,
                              void* d_out, int out_size, void* d_ws, size_t ws_size,
                              hipStream_t stream) {
    const float* feat = (const float*)d_in[0];
    const float* Wq   = (const float*)d_in[1];
    const float* bq   = (const float*)d_in[2];
    const float* Wk   = (const float*)d_in[3];
    const float* bk   = (const float*)d_in[4];
    const float* Wv   = (const float*)d_in[5];
    const float* bv   = (const float*)d_in[6];
    const float* Wo   = (const float*)d_in[7];
    const float* bo   = (const float*)d_in[8];
    const float* W1   = (const float*)d_in[9];
    const float* b1   = (const float*)d_in[10];
    const float* W2   = (const float*)d_in[11];
    const float* b2   = (const float*)d_in[12];
    const float* g1   = (const float*)d_in[13];
    const float* be1  = (const float*)d_in[14];
    const float* g2   = (const float*)d_in[15];
    const float* be2  = (const float*)d_in[16];
    const int*  mol   = (const int*)d_in[17];
    const int*  picked= (const int*)d_in[18];
    float* out = (float*)d_out;

    const int N    = in_sizes[0] / D_EMB;   // total atoms
    const int B    = in_sizes[18];          // molecules
    const int Npad = (N + 127) & ~127;

    char* ws = (char*)d_ws;
    int* offsets = (int*)ws;
    size_t kvOff = (((size_t)(B + 1) * sizeof(int)) + 255) & ~(size_t)255;
    _Float16* kv = (_Float16*)(ws + kvOff);   // [Npad, 2048] f16

    build_offsets<<<(N + 255) / 256, 256, 0, stream>>>(mol, offsets, N, B);
    dim3 gg(Npad / 128, 16, 1);
    kv_gemm<<<gg, 256, 0, stream>>>(feat, Wk, bk, Wv, bv, kv, N);
    attn_tail<<<B, 256, 0, stream>>>(feat, Wq, bq, Wo, bo, W1, b1, W2, b2,
                                     g1, be1, g2, be2, offsets, picked, kv, out, N);
}
